// COMPASSWeightModel_47373489275338
// MI455X (gfx1250) — matrix operand order repaired and checked by reading, not compiled, not run on hardware
//
#include <hip/hip_runtime.h>

#define N_NODES  16384
#define N_EDGESC 262144
#define N_GRAPHS 64
#define HDIM     256
#define KPAD     800   // 3H+4 = 772, padded to multiple of 32

typedef __bf16 bf16_t;
typedef __attribute__((ext_vector_type(16))) __bf16 v16bf;
typedef __attribute__((ext_vector_type(8)))  __bf16 v8bf;
typedef __attribute__((ext_vector_type(8)))  float   v8f;

// ---------- helpers ----------

__device__ __forceinline__ bf16_t to_bf16(float f) {
  union { float f; unsigned u; } in; in.f = f;
  unsigned r = in.u + 0x7FFFu + ((in.u >> 16) & 1u);   // round-to-nearest-even
  union { unsigned short s; bf16_t b; } out;
  out.s = (unsigned short)(r >> 16);
  return out.b;
}

__device__ __forceinline__ v8f wmma_bf16(v16bf a, v16bf b, v8f c) {
  // D = A(16x32) * B(32x16) + C, fp32 accumulate
  return __builtin_amdgcn_wmma_f32_16x16x32_bf16(false, a, false, b, (short)0, c,
                                                 false, false);
}

// A fragment (16x32 bf16) from a row-major [rows][ldk] bf16 matrix.
// ISA layout: lane m=l&15, kh=l>>4; elems 0..7 -> K = kb+kh*8+(0..7),
// elems 8..15 -> K = kb+16+kh*8+(0..7): two contiguous 16B loads.
__device__ __forceinline__ v16bf load_a_rowmajor(const bf16_t* __restrict__ base,
                                                 int ldk, int mrow, int kb, int lane) {
  int m = lane & 15, kh = lane >> 4;
  const bf16_t* p = base + (size_t)(mrow + m) * ldk + kb + kh * 8;
  v8bf c0 = *(const v8bf*)p;
  v8bf c1 = *(const v8bf*)(p + 16);
  v16bf a;
#pragma unroll
  for (int i = 0; i < 8; ++i) { a[i] = c0[i]; a[8 + i] = c1[i]; }
  return a;
}

// B fragment (32x16 bf16) from transposed weights wT[n*ldk + k].
// ISA layout: lane n=l&15, kh=l>>4; elems 0..7 -> K = kb+kh*8+(0..7),
// elems 8..15 -> K = kb+16+kh*8+(0..7).
__device__ __forceinline__ v16bf load_b_wT(const bf16_t* __restrict__ wT,
                                           int ldk, int n0, int kb, int lane) {
  int n = lane & 15, kh = lane >> 4;
  const bf16_t* p = wT + (size_t)(n0 + n) * ldk + kb + kh * 8;
  v8bf c0 = *(const v8bf*)p;
  v8bf c1 = *(const v8bf*)(p + 16);
  v16bf b;
#pragma unroll
  for (int i = 0; i < 8; ++i) { b[i] = c0[i]; b[8 + i] = c1[i]; }
  return b;
}

// ---------- prep kernels ----------

__global__ void k_prep_be(const float* __restrict__ ef, const float* __restrict__ gall,
                          bf16_t* __restrict__ be) {
  int t = blockIdx.x * 256 + threadIdx.x;        // E*256 threads
  int e = t >> 8, k = t & 255;
  float v = (k < 192) ? ef[(size_t)e * 192 + k] : gall[k - 192];
  be[(size_t)t] = to_bf16(v);
}

// w[k*N+n] (fp32) -> wT[n*ldkT+k] (bf16), zero padding for k in [K, ldkT)
__global__ void k_transpose_w(const float* __restrict__ w, bf16_t* __restrict__ wT,
                              int K, int N, int ldkT) {
  int t = blockIdx.x * 256 + threadIdx.x;
  if (t >= N * ldkT) return;
  int n = t / ldkT, k = t % ldkT;
  float v = (k < K) ? w[(size_t)k * N + n] : 0.f;
  wT[(size_t)n * ldkT + k] = to_bf16(v);
}

__global__ void k_zero_nodes(float* __restrict__ bv, unsigned* __restrict__ smax,
                             float* __restrict__ ssum) {
  int n = blockIdx.x, c = threadIdx.x;
  bv[(size_t)n * HDIM + c] = 0.f;
  if (c == 0) { smax[n] = 0u; ssum[n] = 0.f; }
}

// ---------- attention score: sigmoid(relu(b_e@W1+b1)@w2+b2) + segment max ----------

__global__ __launch_bounds__(256) void k_attn(const bf16_t* __restrict__ be,
    const bf16_t* __restrict__ wT1, const float* __restrict__ b1,
    const float* __restrict__ w2, const float* __restrict__ b2,
    const int* __restrict__ rcv, float* __restrict__ score, unsigned* __restrict__ smax) {
  __shared__ float hl[32][HDIM];
  __shared__ float part[32][8];
  int tid = threadIdx.x, lane = tid & 31, wv = tid >> 5, wg = blockIdx.x;
  int mloc = (wv >> 2) * 16, ncol = (wv & 3) * 64;
  int erow = wg * 32 + mloc;

  v8f zero = {0.f, 0.f, 0.f, 0.f, 0.f, 0.f, 0.f, 0.f};
  v8f acc[4];
#pragma unroll
  for (int t = 0; t < 4; ++t) acc[t] = zero;

  for (int kb = 0; kb < HDIM; kb += 32) {
    v16bf a = load_a_rowmajor(be, HDIM, erow, kb, lane);
#pragma unroll
    for (int t = 0; t < 4; ++t) {
      v16bf b = load_b_wT(wT1, HDIM, ncol + t * 16, kb, lane);
      acc[t] = wmma_bf16(a, b, acc[t]);
    }
  }
  int nlo = lane & 15, hh = lane >> 4;
#pragma unroll
  for (int t = 0; t < 4; ++t) {
    int col = ncol + t * 16 + nlo;
    float bias = b1[col];
#pragma unroll
    for (int j = 0; j < 8; ++j) {
      float v = acc[t][j] + bias;
      hl[mloc + hh * 8 + j][col] = v > 0.f ? v : 0.f;
    }
  }
  __syncthreads();
  int el = tid >> 3, ch = tid & 7;
  float p = 0.f;
  for (int i = 0; i < 32; ++i) p += hl[el][ch * 32 + i] * w2[ch * 32 + i];
  part[el][ch] = p;
  __syncthreads();
  if (tid < 32) {
    float s = b2[0];
#pragma unroll
    for (int i = 0; i < 8; ++i) s += part[tid][i];
    float sc = 1.f / (1.f + __expf(-s));       // sigmoid -> (0,1), positive
    int e = wg * 32 + tid;
    score[e] = sc;
    atomicMax(&smax[rcv[e]], __float_as_uint(sc));  // positive floats order as uints
  }
}

// ---------- scatter softmax + aggregation ----------

__global__ void k_expsum(const float* __restrict__ score, const unsigned* __restrict__ smax,
                         const int* __restrict__ rcv, float* __restrict__ exb,
                         float* __restrict__ ssum) {
  int e = blockIdx.x * 256 + threadIdx.x;
  int r = rcv[e];
  float ex = __expf(score[e] - __uint_as_float(smax[r]));
  exb[e] = ex;
  atomicAdd(&ssum[r], ex);
}

__global__ void k_aggregate(const float* __restrict__ exb, const float* __restrict__ ssum,
                            const int* __restrict__ rcv, const bf16_t* __restrict__ be,
                            float* __restrict__ bv) {
  int t = blockIdx.x * 256 + threadIdx.x;   // E*64 threads, 4 cols each
  int e = t >> 6, q = t & 63;
  int r = rcv[e];
  float a = exb[e] / ssum[r];
  const bf16_t* row = be + (size_t)e * HDIM + q * 4;
  float* dst = bv + (size_t)r * HDIM + q * 4;
#pragma unroll
  for (int i = 0; i < 4; ++i) atomicAdd(&dst[i], a * (float)row[i]);
}

__global__ void k_norm(float* __restrict__ bv, bf16_t* __restrict__ bvh,
                       const int* __restrict__ deg) {
  int n = blockIdx.x, c = threadIdx.x;
  float v = bv[(size_t)n * HDIM + c] / (1.f + (float)deg[n]);
  bv[(size_t)n * HDIM + c] = v;
  bvh[(size_t)n * HDIM + c] = to_bf16(v);
}

// ---------- message MLP layer 1: h2 = relu(msg @ eu_w1 + b1), msg built in LDS ----------

__global__ __launch_bounds__(256) void k_msg_gemm(const bf16_t* __restrict__ bvh,
    const bf16_t* __restrict__ be, const bf16_t* __restrict__ wT1,
    const float* __restrict__ b1, const int* __restrict__ srcI,
    const int* __restrict__ rcvI, const int* __restrict__ heads,
    const int* __restrict__ tails, bf16_t* __restrict__ h2) {
  __shared__ bf16_t msg[32][KPAD];   // 50 KB
  int tid = threadIdx.x, lane = tid & 31, wv = tid >> 5, wg = blockIdx.x;

  const bf16_t one = to_bf16(1.f), zval = to_bf16(0.f);
  for (int r = 0; r < 32; ++r) {
    int e = wg * 32 + r;
    int s = srcI[e], rc = rcvI[e];
    for (int k = tid; k < KPAD; k += 256) {
      bf16_t v;
      if (k < 256)        v = bvh[(size_t)s * HDIM + k];
      else if (k == 256)  v = (heads[s >> 8] == s) ? one : zval;
      else if (k == 257)  v = (tails[s >> 8] == s) ? one : zval;
      else if (k < 514)   v = bvh[(size_t)rc * HDIM + (k - 258)];
      else if (k == 514)  v = (heads[rc >> 8] == rc) ? one : zval;
      else if (k == 515)  v = (tails[rc >> 8] == rc) ? one : zval;
      else if (k < 772)   v = be[(size_t)e * HDIM + (k - 516)];
      else                v = zval;
      msg[r][k] = v;
    }
  }
  __syncthreads();

  int mloc = (wv >> 2) * 16, ncol = (wv & 3) * 64;
  v8f zero = {0.f, 0.f, 0.f, 0.f, 0.f, 0.f, 0.f, 0.f};
  v8f acc[4];
#pragma unroll
  for (int t = 0; t < 4; ++t) acc[t] = zero;

  int m = lane & 15, kh = lane >> 4;
  for (int kb = 0; kb < KPAD; kb += 32) {
    const bf16_t* p = &msg[mloc + m][kb + kh * 8];
    v8bf c0 = *(const v8bf*)p;
    v8bf c1 = *(const v8bf*)(p + 16);
    v16bf a;
#pragma unroll
    for (int i = 0; i < 8; ++i) { a[i] = c0[i]; a[8 + i] = c1[i]; }
#pragma unroll
    for (int t = 0; t < 4; ++t) {
      v16bf b = load_b_wT(wT1, KPAD, ncol + t * 16, kb, lane);
      acc[t] = wmma_bf16(a, b, acc[t]);
    }
  }
  int nlo = lane & 15, hg = lane >> 4;
#pragma unroll
  for (int t = 0; t < 4; ++t) {
    int col = ncol + t * 16 + nlo;
    float bias = b1[col];
#pragma unroll
    for (int j = 0; j < 8; ++j) {
      float v = acc[t][j] + bias;
      if (v < 0.f) v = 0.f;
      h2[(size_t)(wg * 32 + mloc + hg * 8 + j) * HDIM + col] = to_bf16(v);
    }
  }
}

// ---------- message MLP layer 2: b_e = h2 @ eu_w2 + b2 (no relu) ----------

__global__ __launch_bounds__(256) void k_edge_gemm(const bf16_t* __restrict__ h2,
    const bf16_t* __restrict__ wT2, const float* __restrict__ b2,
    bf16_t* __restrict__ be) {
  int tid = threadIdx.x, lane = tid & 31, wv = tid >> 5, wg = blockIdx.x;
  int mloc = (wv >> 2) * 16, ncol = (wv & 3) * 64;
  int erow = wg * 32 + mloc;
  v8f zero = {0.f, 0.f, 0.f, 0.f, 0.f, 0.f, 0.f, 0.f};
  v8f acc[4];
#pragma unroll
  for (int t = 0; t < 4; ++t) acc[t] = zero;
  for (int kb = 0; kb < HDIM; kb += 32) {
    v16bf a = load_a_rowmajor(h2, HDIM, erow, kb, lane);
#pragma unroll
    for (int t = 0; t < 4; ++t) {
      v16bf b = load_b_wT(wT2, HDIM, ncol + t * 16, kb, lane);
      acc[t] = wmma_bf16(a, b, acc[t]);
    }
  }
  int nlo = lane & 15, hg = lane >> 4;
#pragma unroll
  for (int t = 0; t < 4; ++t) {
    int col = ncol + t * 16 + nlo;
    float bias = b2[col];
#pragma unroll
    for (int j = 0; j < 8; ++j)
      be[(size_t)(erow + hg * 8 + j) * HDIM + col] = to_bf16(acc[t][j] + bias);
  }
}

// ---------- graph readout ----------

__global__ void k_graphfeat(const float* __restrict__ bv, const int* __restrict__ heads,
                            const int* __restrict__ tails, float* __restrict__ gG) {
  int g = blockIdx.x, c = threadIdx.x;
  float m = -1e30f;
  const float* base = bv + (size_t)g * 256 * HDIM + c;
  for (int i = 0; i < 256; ++i) m = fmaxf(m, base[(size_t)i * HDIM]);
  gG[g * 768 + c]       = m;
  gG[g * 768 + 256 + c] = bv[(size_t)heads[g] * HDIM + c];
  gG[g * 768 + 512 + c] = bv[(size_t)tails[g] * HDIM + c];
}

__global__ __launch_bounds__(256) void k_graph_mlp(const float* __restrict__ gG,
    const float* __restrict__ w1, const float* __restrict__ b1,
    const float* __restrict__ w2, const float* __restrict__ b2,
    float* __restrict__ outG) {
  __shared__ float red[8];
  __shared__ float zl[64];
  __shared__ float wk[64];
  int tid = threadIdx.x, lane = tid & 31, wv = tid >> 5;
  for (int g = 0; g < 64; ++g) {
    float u = b1[tid];
    const float* grow = gG + g * 768;
    for (int k = 0; k < 768; ++k) u += grow[k] * w1[(size_t)k * 256 + tid];
    u = fmaxf(u, 0.f) * w2[tid];
    for (int off = 16; off > 0; off >>= 1) u += __shfl_down(u, off, 32);
    if (lane == 0) red[wv] = u;
    __syncthreads();
    if (tid == 0) {
      float s = b2[0];
#pragma unroll
      for (int i = 0; i < 8; ++i) s += red[i];
      zl[g] = s;
    }
    __syncthreads();
  }
  if (tid == 0) {
    float m = -1e30f;
    for (int g = 0; g < 64; ++g) m = fmaxf(m, zl[g]);
    float s = 0.f;
    for (int g = 0; g < 64; ++g) { float e = __expf(zl[g] - m); wk[g] = e; s += e; }
    for (int g = 0; g < 64; ++g) wk[g] /= s;
  }
  __syncthreads();
  for (int c = tid; c < 768; c += 256) {
    float s = 0.f;
    for (int g = 0; g < 64; ++g) s += wk[g] * gG[g * 768 + c];
    outG[c] = s;
  }
}

__global__ void k_we(const bf16_t* __restrict__ be, const float* __restrict__ ew,
                     const float* __restrict__ eb, float* __restrict__ outW) {
  int e = blockIdx.x * 256 + threadIdx.x;
  const bf16_t* row = be + (size_t)e * HDIM;
  float s = eb[0];
  for (int k = 0; k < HDIM; ++k) s += (float)row[k] * ew[k];
  outW[e] = 1.f / (1.f + __expf(-s));
}

// ---------- launch ----------

extern "C" void kernel_launch(void* const* d_in, const int* in_sizes, int n_in,
                              void* d_out, int out_size, void* d_ws, size_t ws_size,
                              hipStream_t stream) {
  const float* ef    = (const float*)d_in[0];
  const float* gall  = (const float*)d_in[1];
  const float* ea_w1 = (const float*)d_in[2];
  const float* ea_b1 = (const float*)d_in[3];
  const float* ea_w2 = (const float*)d_in[4];
  const float* ea_b2 = (const float*)d_in[5];
  const float* eu_w1 = (const float*)d_in[6];
  const float* eu_b1 = (const float*)d_in[7];
  const float* eu_w2 = (const float*)d_in[8];
  const float* eu_b2 = (const float*)d_in[9];
  const float* gw_w1 = (const float*)d_in[10];
  const float* gw_b1 = (const float*)d_in[11];
  const float* gw_w2 = (const float*)d_in[12];
  const float* gw_b2 = (const float*)d_in[13];
  const float* ew_w  = (const float*)d_in[14];
  const float* ew_b  = (const float*)d_in[15];
  const int* eidx  = (const int*)d_in[16];
  const int* deg   = (const int*)d_in[17];
  const int* heads = (const int*)d_in[19];
  const int* tails = (const int*)d_in[20];
  const int* srcI = eidx;
  const int* rcvI = eidx + N_EDGESC;

  char* ws = (char*)d_ws;
  size_t off = 0;
  auto carve = [&](size_t bytes) -> void* {
    void* p = (void*)(ws + off);
    off += (bytes + 255) & ~(size_t)255;
    return p;
  };
  bf16_t*   be    = (bf16_t*)carve((size_t)N_EDGESC * HDIM * 2);
  bf16_t*   h2    = (bf16_t*)carve((size_t)N_EDGESC * HDIM * 2);
  bf16_t*   wTea1 = (bf16_t*)carve((size_t)HDIM * HDIM * 2);
  bf16_t*   wTeu1 = (bf16_t*)carve((size_t)HDIM * KPAD * 2);
  bf16_t*   wTeu2 = (bf16_t*)carve((size_t)HDIM * HDIM * 2);
  float*    score = (float*)carve((size_t)N_EDGESC * 4);
  float*    exb   = (float*)carve((size_t)N_EDGESC * 4);
  unsigned* smax  = (unsigned*)carve((size_t)N_NODES * 4);
  float*    ssum  = (float*)carve((size_t)N_NODES * 4);
  float*    bv    = (float*)carve((size_t)N_NODES * HDIM * 4);
  bf16_t*   bvh   = (bf16_t*)carve((size_t)N_NODES * HDIM * 2);
  float*    gG    = (float*)carve((size_t)N_GRAPHS * 768 * 4);

  float* outW = (float*)d_out;        // w_e [E]
  float* outG = outW + N_EDGESC;      // g_all_out [768]

  k_prep_be<<<N_EDGESC, 256, 0, stream>>>(ef, gall, be);
  k_transpose_w<<<(HDIM * HDIM + 255) / 256, 256, 0, stream>>>(ea_w1, wTea1, HDIM, HDIM, HDIM);
  k_transpose_w<<<(HDIM * KPAD + 255) / 256, 256, 0, stream>>>(eu_w1, wTeu1, 772, HDIM, KPAD);
  k_transpose_w<<<(HDIM * HDIM + 255) / 256, 256, 0, stream>>>(eu_w2, wTeu2, HDIM, HDIM, HDIM);

  for (int it = 0; it < 3; ++it) {
    k_zero_nodes<<<N_NODES, 256, 0, stream>>>(bv, smax, ssum);
    k_attn<<<N_EDGESC / 32, 256, 0, stream>>>(be, wTea1, ea_b1, ea_w2, ea_b2, rcvI, score, smax);
    k_expsum<<<N_EDGESC / 256, 256, 0, stream>>>(score, smax, rcvI, exb, ssum);
    k_aggregate<<<N_EDGESC / 4, 256, 0, stream>>>(exb, ssum, rcvI, be, bv);
    k_norm<<<N_NODES, 256, 0, stream>>>(bv, bvh, deg);
    k_msg_gemm<<<N_EDGESC / 32, 256, 0, stream>>>(bvh, be, wTeu1, eu_b1, srcI, rcvI, heads, tails, h2);
    k_edge_gemm<<<N_EDGESC / 32, 256, 0, stream>>>(h2, wTeu2, eu_b2, be);
  }
  k_graphfeat<<<N_GRAPHS, 256, 0, stream>>>(bv, heads, tails, gG);
  k_graph_mlp<<<1, 256, 0, stream>>>(gG, gw_w1, gw_b1, gw_w2, gw_b2, outG);
  k_we<<<N_EDGESC / 256, 256, 0, stream>>>(be, ew_w, ew_b, outW);
}
